// GRU_56401510531141
// MI455X (gfx1250) — matrix operand order repaired and checked by reading, not compiled, not run on hardware
//
#include <hip/hip_runtime.h>
#include <hip/hip_bf16.h>
#include <stdint.h>

// ---------------------------------------------------------------------------
// GRU (T=8192, I=512, H=2048, O=512) for MI455X / gfx1250 (wave32, WMMA).
//
// Strategy:
//   K_pack_*: convert weights/x to bf16; recurrent + input weights are
//             pre-swizzled into a 16x32 bf16 A-fragment lane layout
//             (contiguous k = 16*(l/16)+s, as B) so the hot loops load fragments with b128 loads.
//   K_gemm  : parallel WMMA GEMM P[g][t][j] = b_g[j] + Wg_x . x_t  (bf16 MACs,
//             fp32 accum). 51.5 GFLOP fully parallel -> trivial on WMMA.
//   K_recur : 64 persistent WGs, 8 waves each. Per step:
//             phase A: r,z slices via WMMA matvec (B = h broadcast into all
//                      16 columns), split-K over 2 waves, ds_add_f32 reduce;
//             publish r*h (bf16) -> global, grid barrier;
//             phase B: h_cand slice via WMMA matvec on r*h; h' update; publish
//                      h (bf16) -> global, grid barrier.
//             h fp32 slice lives in LDS for the whole kernel. Broadcast
//             vectors staged into LDS with global_load_async_to_lds_b128.
//   K_out   : out = Wo @ h_final + bo (tiny, fp32 VALU).
// ---------------------------------------------------------------------------

typedef __attribute__((ext_vector_type(16))) __bf16 v16bf;
typedef __attribute__((ext_vector_type(8)))  float  v8f;

#define T_LEN 8192
#define IN    512
#define H     2048
#define OUTN  512
#define NWG   64
#define SLICE 32            // H / NWG rows per gate per WG

// workspace byte offsets (all 256B aligned)
#define OFF_WH   ((size_t)0)                         // bf16 3*2048*2048   = 25165824 B
#define OFF_WX   ((size_t)25165824)                  // bf16 3*2048*512    =  6291456 B
#define OFF_XBF  ((size_t)31457280)                  // bf16 8192*512      =  8388608 B
#define OFF_P    ((size_t)39845888)                  // f32  3*8192*2048   = 201326592 B
#define OFF_HBF  ((size_t)241172480)                 // bf16 2048
#define OFF_RHBF ((size_t)241176576)                 // bf16 2048
#define OFF_HF   ((size_t)241180672)                 // f32  2048
#define OFF_BAR  ((size_t)241188864)                 // 2 x u32
// total ~241.2 MB of d_ws required.

// ---------------------------------------------------------------------------
// pack kernels
// ---------------------------------------------------------------------------
__global__ void k_pack_x(const float* __restrict__ x, __bf16* __restrict__ xbf) {
  size_t i = (size_t)blockIdx.x * blockDim.x + threadIdx.x;
  if (i < (size_t)T_LEN * IN) xbf[i] = (__bf16)x[i];
}

// A-fragment (16x32 bf16, contiguous k order, same as B): lane l holds row (l%16);
// bytes 0..15  = K in [k0 + (l/16)*16,     +8)
// bytes 16..31 = K in [k0 + 8 + (l/16)*16, +8)
__global__ void k_pack_wh(const float* __restrict__ Wr, const float* __restrict__ Wz,
                          const float* __restrict__ Wh, __bf16* __restrict__ whfrag) {
  unsigned tid = blockIdx.x * blockDim.x + threadIdx.x;     // 3*128*64*32
  unsigned lane = tid & 31, kt = (tid >> 5) & 63, jt = (tid >> 11) & 127, g = tid >> 18;
  if (g >= 3) return;
  const float* W = (g == 0) ? Wr : (g == 1) ? Wz : Wh;
  unsigned row = jt * 16 + (lane & 15);
  unsigned k0  = IN + kt * 32 + (lane >> 4) * 16;           // hidden cols start at IN
  __bf16* dst = whfrag + (((size_t)g * 128 + jt) * 64 + kt) * 512 + (size_t)lane * 16;
  const float* src = W + (size_t)row * (IN + H);
#pragma unroll
  for (int e = 0; e < 8; ++e) dst[e] = (__bf16)src[k0 + e];
#pragma unroll
  for (int e = 0; e < 8; ++e) dst[8 + e] = (__bf16)src[k0 + 8 + e];
}

__global__ void k_pack_wx(const float* __restrict__ Wr, const float* __restrict__ Wz,
                          const float* __restrict__ Wh, __bf16* __restrict__ wxfrag) {
  unsigned tid = blockIdx.x * blockDim.x + threadIdx.x;     // 3*128*16*32
  unsigned lane = tid & 31, it = (tid >> 5) & 15, jt = (tid >> 9) & 127, g = tid >> 16;
  if (g >= 3) return;
  const float* W = (g == 0) ? Wr : (g == 1) ? Wz : Wh;
  unsigned row = jt * 16 + (lane & 15);
  unsigned k0  = it * 32 + (lane >> 4) * 16;
  __bf16* dst = wxfrag + (((size_t)g * 128 + jt) * 16 + it) * 512 + (size_t)lane * 16;
  const float* src = W + (size_t)row * (IN + H);
#pragma unroll
  for (int e = 0; e < 8; ++e) dst[e] = (__bf16)src[k0 + e];
#pragma unroll
  for (int e = 0; e < 8; ++e) dst[8 + e] = (__bf16)src[k0 + 8 + e];
}

__global__ void k_init(__bf16* hbf, __bf16* rhbf, float* hfinal, unsigned* bar) {
  int t = threadIdx.x;
  for (int i = t; i < H; i += 256) {
    hbf[i] = (__bf16)0.0f; rhbf[i] = (__bf16)0.0f; hfinal[i] = 0.0f;
  }
  if (t < 2) bar[t] = 0u;
}

// ---------------------------------------------------------------------------
// parallel GEMM: P[g][t][j] = bias_g[j] + sum_i Wg_x[j,i] * x[t,i]
// one wave computes one 16(j) x 16(t) tile, K=512 (16 WMMAs).
// ---------------------------------------------------------------------------
__global__ void __launch_bounds__(256) k_gemm(const __bf16* __restrict__ wxfrag,
                                              const __bf16* __restrict__ xbf,
                                              const float* __restrict__ br,
                                              const float* __restrict__ bz,
                                              const float* __restrict__ bh,
                                              float* __restrict__ P) {
  unsigned wave = threadIdx.x >> 5, lane = threadIdx.x & 31;
  unsigned tile = blockIdx.x * 8 + wave;                    // 3*512*128 tiles
  unsigned jt = tile & 127;
  unsigned tt = (tile >> 7) & 511;
  unsigned g  = tile >> 16;
  const float* bias = (g == 0) ? br : (g == 1) ? bz : bh;
  unsigned t = tt * 16 + (lane & 15);                       // B column -> time
  const __bf16* afrag = wxfrag + (((size_t)g * 128 + jt) * 16) * 512 + (size_t)lane * 16;
  const __bf16* brow  = xbf + (size_t)t * IN + (lane >> 4) * 16;
  v8f c = (v8f)0.0f;
#pragma unroll
  for (int it = 0; it < 16; ++it) {
    v16bf a = *(const v16bf*)(afrag + (size_t)it * 512);
    v16bf b = *(const v16bf*)(brow + it * 32);              // B[kk,n] = x[t0+n, i0+kk]
    c = __builtin_amdgcn_wmma_f32_16x16x32_bf16(false, a, false, b, (short)0, c, false, false);
  }
  unsigned jbase = jt * 16 + (lane >> 4) * 8;               // rows v + 8*(lane/16)
  float* dst = P + ((size_t)g * T_LEN + t) * H + jbase;
#pragma unroll
  for (int v = 0; v < 8; ++v) dst[v] = c[v] + bias[jbase + v];
}

// ---------------------------------------------------------------------------
// grid barrier (generation counting; re-initialized every launch by k_init)
// ---------------------------------------------------------------------------
__device__ __forceinline__ void grid_barrier(unsigned* bar, unsigned nwg, unsigned& my_gen) {
  __threadfence();                                          // release our stores
  __syncthreads();
  if (threadIdx.x == 0) {
    unsigned arrived = __hip_atomic_fetch_add(&bar[0], 1u, __ATOMIC_ACQ_REL, __HIP_MEMORY_SCOPE_AGENT);
    if (arrived == nwg - 1) {
      __hip_atomic_store(&bar[0], 0u, __ATOMIC_RELAXED, __HIP_MEMORY_SCOPE_AGENT);
      __hip_atomic_store(&bar[1], my_gen + 1u, __ATOMIC_RELEASE, __HIP_MEMORY_SCOPE_AGENT);
    } else {
      while (__hip_atomic_load(&bar[1], __ATOMIC_ACQUIRE, __HIP_MEMORY_SCOPE_AGENT) == my_gen)
        __builtin_amdgcn_s_sleep(1);
    }
  }
  my_gen++;
  __syncthreads();
  __threadfence();                                          // acquire: invalidate stale lines
}

// ---------------------------------------------------------------------------
// persistent recurrence kernel: 64 WGs x 256 threads (8 waves)
// ---------------------------------------------------------------------------
__global__ void __launch_bounds__(256) k_recur(const __bf16* __restrict__ whfrag,
                                               const float* __restrict__ P,
                                               __bf16* __restrict__ g_hbf,
                                               __bf16* __restrict__ g_rhbf,
                                               float* __restrict__ hfinal,
                                               unsigned* bar) {
  __shared__ __align__(64) __bf16 sh_hbf[H];     // broadcast h   (bf16, staged per step)
  __shared__ __align__(64) __bf16 sh_rhbf[H];    // broadcast r*h (bf16, staged per step)
  __shared__ float sh_h[SLICE];                  // persistent fp32 h slice
  __shared__ float sh_z[SLICE];
  __shared__ float sh_acc[3][SLICE];             // r / z / h_cand pre-activations

  const unsigned tid  = threadIdx.x;
  const unsigned wave = tid >> 5, lane = tid & 31;
  const unsigned wg   = blockIdx.x;
  const unsigned j0   = wg * SLICE;
  unsigned my_gen = 0;

  if (tid < SLICE) sh_h[tid] = 0.0f;             // h0 = 0
  __syncthreads();

  // phase A: 8 waves = (gate r/z) x (j-tile 0/1) x (K half 0/1), 32 k-tiles each
  const unsigned gaA = wave & 1, jtA = (wave >> 1) & 1, khA = wave >> 2;
  // phase B: 8 waves = (j-tile 0/1) x (K quarter 0..3), 16 k-tiles each
  const unsigned jtB = wave & 1, kqB = wave >> 1;
  const __bf16* afragA = whfrag + (((size_t)gaA * 128 + (wg * 2 + jtA)) * 64 + (size_t)khA * 32) * 512
                                + (size_t)lane * 16;
  const __bf16* afragB = whfrag + (((size_t)2   * 128 + (wg * 2 + jtB)) * 64 + (size_t)kqB * 16) * 512
                                + (size_t)lane * 16;
  const unsigned bhalf = (lane >> 4) * 16;       // which 16 K-values this half-wave holds

  for (int t = 0; t < T_LEN; ++t) {
    // ---- stage full h (bf16) into LDS via async DMA; load P slices ----
    {
      unsigned dst = (unsigned)(uintptr_t)(&sh_hbf[tid * 8]);
      const __bf16* src = g_hbf + tid * 8;
      asm volatile("global_load_async_to_lds_b128 %0, %1, off" :: "v"(dst), "v"(src) : "memory");
    }
    if (tid < 96) {
      unsigned gg = tid >> 5, j = tid & 31;
      sh_acc[gg][j] = P[((size_t)gg * T_LEN + t) * H + j0 + j];
    }
    asm volatile("s_wait_asynccnt 0" ::: "memory");
    __syncthreads();

    // ---- phase A: acc_{r,z} += Wg_h[slice,:] @ h  (WMMA matvec, B replicated) ----
    {
      v8f c = (v8f)0.0f;
#pragma unroll 4
      for (int kt = 0; kt < 32; ++kt) {
        v16bf a = *(const v16bf*)(afragA + (size_t)kt * 512);
        v16bf b = *(const v16bf*)(&sh_hbf[(khA * 32 + kt) * 32 + bhalf]);
        c = __builtin_amdgcn_wmma_f32_16x16x32_bf16(false, a, false, b, (short)0, c, false, false);
      }
      if ((lane & 15) == 0) {                    // lanes 0,16 hold rows 0..7 / 8..15
        unsigned r0 = jtA * 16 + (lane >> 4) * 8;
#pragma unroll
        for (int v = 0; v < 8; ++v) atomicAdd(&sh_acc[gaA][r0 + v], c[v]);
      }
    }
    __syncthreads();

    // ---- gates: r -> publish r*h (bf16); z -> keep in LDS ----
    if (tid < SLICE) {
      float r = 1.0f / (1.0f + __expf(-sh_acc[0][tid]));
      g_rhbf[j0 + tid] = (__bf16)(r * sh_h[tid]);
    } else if (tid < 2 * SLICE) {
      unsigned j = tid - SLICE;
      sh_z[j] = 1.0f / (1.0f + __expf(-sh_acc[1][j]));
    }
    grid_barrier(bar, NWG, my_gen);

    // ---- stage full r*h (bf16) into LDS ----
    {
      unsigned dst = (unsigned)(uintptr_t)(&sh_rhbf[tid * 8]);
      const __bf16* src = g_rhbf + tid * 8;
      asm volatile("global_load_async_to_lds_b128 %0, %1, off" :: "v"(dst), "v"(src) : "memory");
      asm volatile("s_wait_asynccnt 0" ::: "memory");
    }
    __syncthreads();

    // ---- phase B: acc_h += Wh_h[slice,:] @ (r*h) ----
    {
      v8f c = (v8f)0.0f;
#pragma unroll 4
      for (int kt = 0; kt < 16; ++kt) {
        v16bf a = *(const v16bf*)(afragB + (size_t)kt * 512);
        v16bf b = *(const v16bf*)(&sh_rhbf[(kqB * 16 + kt) * 32 + bhalf]);
        c = __builtin_amdgcn_wmma_f32_16x16x32_bf16(false, a, false, b, (short)0, c, false, false);
      }
      if ((lane & 15) == 0) {
        unsigned r0 = jtB * 16 + (lane >> 4) * 8;
#pragma unroll
        for (int v = 0; v < 8; ++v) atomicAdd(&sh_acc[2][r0 + v], c[v]);
      }
    }
    __syncthreads();

    // ---- h' = z*h + (1-z)*tanh(acc_h); publish bf16 h ----
    if (tid < SLICE) {
      float hc = tanhf(sh_acc[2][tid]);
      float z  = sh_z[tid];
      float hn = z * sh_h[tid] + (1.0f - z) * hc;
      sh_h[tid] = hn;
      g_hbf[j0 + tid] = (__bf16)hn;
    }
    grid_barrier(bar, NWG, my_gen);
  }

  if (tid < SLICE) hfinal[j0 + tid] = sh_h[tid];
}

// ---------------------------------------------------------------------------
// output projection: out = Wo @ h_final + bo  (fp32, exact)
// ---------------------------------------------------------------------------
__global__ void k_out(const float* __restrict__ Wo, const float* __restrict__ bo,
                      const float* __restrict__ hfinal, float* __restrict__ out) {
  unsigned o = blockIdx.x * blockDim.x + threadIdx.x;
  if (o >= OUTN) return;
  float acc = bo[o];
  const float* w = Wo + (size_t)o * H;
  for (int k = 0; k < H; ++k) acc = fmaf(w[k], hfinal[k], acc);
  out[o] = acc;
}

// ---------------------------------------------------------------------------
extern "C" void kernel_launch(void* const* d_in, const int* in_sizes, int n_in,
                              void* d_out, int out_size, void* d_ws, size_t ws_size,
                              hipStream_t stream) {
  (void)in_sizes; (void)n_in; (void)out_size; (void)ws_size;
  const float* x  = (const float*)d_in[0];
  const float* Wr = (const float*)d_in[1];
  const float* br = (const float*)d_in[2];
  const float* Wz = (const float*)d_in[3];
  const float* bz = (const float*)d_in[4];
  const float* Wh = (const float*)d_in[5];
  const float* bh = (const float*)d_in[6];
  const float* Wo = (const float*)d_in[7];
  const float* bo = (const float*)d_in[8];

  char* ws = (char*)d_ws;
  __bf16*   whfrag = (__bf16*)(ws + OFF_WH);
  __bf16*   wxfrag = (__bf16*)(ws + OFF_WX);
  __bf16*   xbf    = (__bf16*)(ws + OFF_XBF);
  float*    P      = (float*) (ws + OFF_P);
  __bf16*   hbf    = (__bf16*)(ws + OFF_HBF);
  __bf16*   rhbf   = (__bf16*)(ws + OFF_RHBF);
  float*    hfin   = (float*) (ws + OFF_HF);
  unsigned* bar    = (unsigned*)(ws + OFF_BAR);

  k_pack_x <<<(T_LEN * IN) / 256, 256, 0, stream>>>(x, xbf);
  k_pack_wh<<<3072, 256, 0, stream>>>(Wr, Wz, Wh, whfrag);
  k_pack_wx<<<768,  256, 0, stream>>>(Wr, Wz, Wh, wxfrag);
  k_init   <<<1,    256, 0, stream>>>(hbf, rhbf, hfin, bar);
  k_gemm   <<<24576, 256, 0, stream>>>(wxfrag, xbf, br, bz, bh, P);
  k_recur  <<<NWG,  256, 0, stream>>>(whfrag, P, hbf, rhbf, hfin, bar);
  k_out    <<<2,    256, 0, stream>>>(Wo, bo, hfin, (float*)d_out);
}
